// DecoderRNN_DTP_ONLY_15719580303480
// MI455X (gfx1250) — matrix operand order repaired and checked by reading, not compiled, not run on hardware
//
#include <hip/hip_runtime.h>
#include <math.h>

// ---------------- problem constants ----------------
#define HB   512      // hidden size H
#define BB   2048     // batch
#define EE   2048     // encoder input features
#define G3   1536     // 3*H gate width
#define NSTEPS 60

typedef _Float16 v16h __attribute__((ext_vector_type(16)));
typedef _Float16 v8h  __attribute__((ext_vector_type(8)));
typedef float    v8f  __attribute__((ext_vector_type(8)));

union F8 { float4 q[2]; float f[8]; };

// ---------------- WMMA helpers (CDNA5 gfx1250, wave32) ----------------
// A operand, 16x32 f16, row-major source with row stride ld (elements).
// ISA layout: lanes 0-15: M=lane, halves = K {k0..k0+7, k0+16..k0+23};
//             lanes 16-31: M=lane-16, halves = K {k0+8..+15, k0+24..+31}.
__device__ __forceinline__ v16h load_a16(const _Float16* p, int ld) {
    const int lane = threadIdx.x & 31;
    const int hl = lane & 15, hi = lane >> 4;
    const _Float16* r = p + (size_t)hl * ld + hi * 8;
    union { v16h v; v8h h[2]; } u;
    u.h[0] = *reinterpret_cast<const v8h*>(r);
    u.h[1] = *reinterpret_cast<const v8h*>(r + 16);
    return u.v;
}

// B operand, 32x16 f16, from k-major (transposed) weights: W^T[k][n], row stride ld.
// ISA layout: lane l holds N = n0 + l%16, K {k0+8h..+7, k0+16+8h..+7}, h = l/16.
__device__ __forceinline__ v16h load_b16(const _Float16* p, int ld) {
    const int lane = threadIdx.x & 31;
    const int hl = lane & 15, hi = lane >> 4;
    const _Float16* r = p + (size_t)(hi * 8) * ld + hl;
    v16h v;
#pragma unroll
    for (int s = 0; s < 8; ++s) {
        v[s] = r[(size_t)s * ld];
        v[s + 8] = r[(size_t)(s + 16) * ld];
    }
    return v;
}

__device__ __forceinline__ v8f wmma16(v16h a, v16h b, v8f c) {
    // D = A(16x32) * B(32x16) + C, f32 accumulate
    return __builtin_amdgcn_wmma_f32_16x16x32_f16(
        false, a, false, b, (short)0, c, false, false);
}

__device__ __forceinline__ float sigmoidf_fast(float x) {
    return 1.0f / (1.0f + __expf(-x));
}

// ---------------- prep kernels ----------------
__global__ void k_cast16(_Float16* __restrict__ dst, const float* __restrict__ src, int n) {
    for (int i = blockIdx.x * blockDim.x + threadIdx.x; i < n; i += gridDim.x * blockDim.x)
        dst[i] = (_Float16)src[i];
}

// src: [rows][cols] f32 row-major -> dst: [cols][rows] f16 (k-major weights)
__global__ void k_transpose16(_Float16* __restrict__ dst, const float* __restrict__ src,
                              int rows, int cols) {
    const int n = rows * cols;
    for (int i = blockIdx.x * blockDim.x + threadIdx.x; i < n; i += gridDim.x * blockDim.x) {
        const int r = i / cols, c = i % cols;
        dst[(size_t)c * rows + r] = (_Float16)src[i];
    }
}

// ---------------- encoder: x = relu(dtp @ W_enc^T + b_enc), f16 out ----------------
__global__ __launch_bounds__(256)
void k_encoder(const _Float16* __restrict__ dtp16, const _Float16* __restrict__ wencT,
               const float* __restrict__ b_enc, _Float16* __restrict__ x16) {
    const int row0 = blockIdx.x * 16;
    const int wave = threadIdx.x >> 5, lane = threadIdx.x & 31;
    const int hl = lane & 15, hi = lane >> 4;

    v8f acc[4] = {};
    for (int ks = 0; ks < EE / 32; ++ks) {
        const int k0 = ks * 32;
        const v16h a = load_a16(dtp16 + (size_t)row0 * EE + k0, EE);
#pragma unroll
        for (int t = 0; t < 4; ++t) {
            const int n0 = wave * 64 + t * 16;
            const v16h b = load_b16(wencT + (size_t)k0 * HB + n0, HB);
            acc[t] = wmma16(a, b, acc[t]);
        }
    }
#pragma unroll
    for (int t = 0; t < 4; ++t) {
        const int n = wave * 64 + t * 16 + hl;
        const float bb = b_enc[n];
#pragma unroll
        for (int v = 0; v < 8; ++v) {
            const int m = v + hi * 8;
            float val = acc[t][v] + bb;
            val = val > 0.0f ? val : 0.0f;
            x16[(size_t)(row0 + m) * HB + n] = (_Float16)val;
        }
    }
}

// ---- giT[l][n][b] = (x @ W_ih_l^T + b_ih_l)[b][n]  (step-invariant, gate-major) ----
// Transposed layout makes the per-step gi reads in k_rnn contiguous in the
// batch dim (matches the C-tile VGPR layout) -> float4 loads instead of b32.
__global__ __launch_bounds__(256)
void k_gi(const _Float16* __restrict__ x16, const _Float16* __restrict__ wihT_all,
          const float* __restrict__ bi1, const float* __restrict__ bi2,
          const float* __restrict__ bi3, float* __restrict__ giT_all) {
    const int layer = blockIdx.y;
    const _Float16* wT = wihT_all + (size_t)layer * HB * G3;
    const float* bi = (layer == 0) ? bi1 : ((layer == 1) ? bi2 : bi3);
    float* giT = giT_all + (size_t)layer * G3 * BB;

    const int row0 = blockIdx.x * 16;
    const int wave = threadIdx.x >> 5, lane = threadIdx.x & 31;
    const int hl = lane & 15, hi = lane >> 4;

    v8f acc[12] = {};
    for (int ks = 0; ks < HB / 32; ++ks) {
        const int k0 = ks * 32;
        const v16h a = load_a16(x16 + (size_t)row0 * HB + k0, HB);
#pragma unroll
        for (int t = 0; t < 12; ++t) {
            const int n0 = wave * 192 + t * 16;
            const v16h b = load_b16(wT + (size_t)k0 * G3 + n0, G3);
            acc[t] = wmma16(a, b, acc[t]);
        }
    }
#pragma unroll
    for (int t = 0; t < 12; ++t) {
        const int n = wave * 192 + t * 16 + hl;
        const float bb = bi[n];
        F8 o8;
#pragma unroll
        for (int v = 0; v < 8; ++v) o8.f[v] = acc[t][v] + bb;
        // element (m = v + hi*8, n) -> giT[n][row0 + hi*8 + v]: contiguous in v
        float4* p = (float4*)(giT + (size_t)n * BB + row0 + hi * 8);
        p[0] = o8.q[0];
        p[1] = o8.q[1];
    }
}

// ---------------- persistent fused 3-layer GRU, 60 steps ----------------
// One workgroup owns 16 batch rows for the whole sequence; h lives in LDS.
// Wave w owns hidden columns [w*64, w*64+64) and computes r/z/n gate tiles for
// those columns itself -> gate fusion is pure register math (no shuffles).
__global__ __launch_bounds__(256)
void k_rnn(const _Float16* __restrict__ whhT_all, const float* __restrict__ giT_all,
           const float* __restrict__ bh1, const float* __restrict__ bh2,
           const float* __restrict__ bh3, const float* __restrict__ W_out,
           const float* __restrict__ b_out, float* __restrict__ out) {
    __shared__ _Float16 h16[16][HB + 8];   // f16 copy used as WMMA A operand
    __shared__ float    h32[16][HB];       // f32 master state
    __shared__ float    wout_s[4][HB];     // output projection weights

    const int row0 = blockIdx.x * 16;
    const int tid = threadIdx.x;
    const int wave = tid >> 5, lane = tid & 31;
    const int hl = lane & 15, hi = lane >> 4;

    for (int i = tid; i < 16 * HB; i += 256) {
        h32[i / HB][i % HB] = 0.0f;
        h16[i / HB][i % HB] = (_Float16)0.0f;
    }
    for (int i = tid; i < 4 * HB; i += 256) wout_s[i >> 9][i & (HB - 1)] = W_out[i];
    __syncthreads();

    // hidden-side biases are step-invariant: preload per lane
    const float* bhs[3] = { bh1, bh2, bh3 };
    float bh[3][3][4];
    for (int l = 0; l < 3; ++l)
#pragma unroll
        for (int g = 0; g < 3; ++g)
#pragma unroll
            for (int ct = 0; ct < 4; ++ct)
                bh[l][g][ct] = bhs[l][g * HB + wave * 64 + ct * 16 + hl];

    const float bo = b_out[(tid >> 6) & 3];  // b_out[o] for projection (tid<256)

    for (int t = 0; t < NSTEPS; ++t) {
        for (int l = 0; l < 3; ++l) {
            const _Float16* wT = whhT_all + (size_t)l * HB * G3;
            const float* giT = giT_all + (size_t)l * G3 * BB;

            v8f acc[12] = {};  // [g*4 + ct]
            for (int ks = 0; ks < HB / 32; ++ks) {
                const int k0 = ks * 32;
                const v16h a = load_a16(&h16[0][0] + k0, HB + 8);
                if (ks + 1 < HB / 32)  // hint next K-slab of weights into cache
                    __builtin_prefetch(wT + (size_t)(k0 + 32 + lane) * G3 + wave * 192, 0, 3);
#pragma unroll
                for (int g = 0; g < 3; ++g)
#pragma unroll
                    for (int ct = 0; ct < 4; ++ct) {
                        const int n0 = g * HB + wave * 64 + ct * 16;
                        const v16h b = load_b16(wT + (size_t)k0 * G3 + n0, G3);
                        acc[g * 4 + ct] = wmma16(a, b, acc[g * 4 + ct]);
                    }
            }
            // GRU gate fusion + state update for this wave's 64 columns
#pragma unroll
            for (int ct = 0; ct < 4; ++ct) {
                const int c = wave * 64 + ct * 16 + hl;
                // gi for 8 consecutive batch rows (m = hi*8 + v): two float4 each
                F8 gr8, gz8, gn8;
                {
                    const float4* pr = (const float4*)(giT + (size_t)(0 * HB + c) * BB + row0 + hi * 8);
                    const float4* pz = (const float4*)(giT + (size_t)(1 * HB + c) * BB + row0 + hi * 8);
                    const float4* pn = (const float4*)(giT + (size_t)(2 * HB + c) * BB + row0 + hi * 8);
                    gr8.q[0] = pr[0]; gr8.q[1] = pr[1];
                    gz8.q[0] = pz[0]; gz8.q[1] = pz[1];
                    gn8.q[0] = pn[0]; gn8.q[1] = pn[1];
                }
#pragma unroll
                for (int v = 0; v < 8; ++v) {
                    const int m = v + hi * 8;
                    const float gr = gr8.f[v] + acc[0 * 4 + ct][v] + bh[l][0][ct];
                    const float gz = gz8.f[v] + acc[1 * 4 + ct][v] + bh[l][1][ct];
                    const float hn = acc[2 * 4 + ct][v] + bh[l][2][ct];
                    const float r = sigmoidf_fast(gr);
                    const float z = sigmoidf_fast(gz);
                    const float n = tanhf(gn8.f[v] + r * hn);
                    const float hold = h32[m][c];
                    const float hnew = (1.0f - z) * n + z * hold;
                    h32[m][c] = hnew;
                    h16[m][c] = (_Float16)hnew;
                }
            }
            __syncthreads();
        }
        // output projection: out[b][o*60 + t] = h . W_out[o] + b_out[o]
        // all 256 threads: 4 partial sums per (m,o) pair, shuffle-reduce in-wave
        {
            const int pair = tid >> 2;          // 0..63 : (o,m)
            const int part = tid & 3;           // 0..3  : K quarter
            const int m = pair & 15, o = pair >> 4;
            float s = 0.0f;
            const int k0 = part * (HB / 4);
#pragma unroll 4
            for (int k = k0; k < k0 + HB / 4; ++k) s += h32[m][k] * wout_s[o][k];
            s += __shfl_xor(s, 1, 32);
            s += __shfl_xor(s, 2, 32);
            if (part == 0)
                out[(size_t)(row0 + m) * (4 * NSTEPS) + o * NSTEPS + t] = s + bo;
        }
        __syncthreads();  // projection reads h before next step's update writes it
    }
}

// ---------------- host launcher ----------------
extern "C" void kernel_launch(void* const* d_in, const int* in_sizes, int n_in,
                              void* d_out, int out_size, void* d_ws, size_t ws_size,
                              hipStream_t stream) {
    (void)in_sizes; (void)n_in; (void)out_size; (void)ws_size;
    // d_in order: context(unused), dtp_features, W_enc, b_enc, W_out, b_out,
    //             then per layer l=1..3: W_ih, W_hh, b_ih, b_hh
    const float* dtp   = (const float*)d_in[1];
    const float* W_enc = (const float*)d_in[2];
    const float* b_enc = (const float*)d_in[3];
    const float* W_out = (const float*)d_in[4];
    const float* b_out = (const float*)d_in[5];
    const float* W_ih[3] = { (const float*)d_in[6],  (const float*)d_in[10], (const float*)d_in[14] };
    const float* W_hh[3] = { (const float*)d_in[7],  (const float*)d_in[11], (const float*)d_in[15] };
    const float* b_ih[3] = { (const float*)d_in[8],  (const float*)d_in[12], (const float*)d_in[16] };
    const float* b_hh[3] = { (const float*)d_in[9],  (const float*)d_in[13], (const float*)d_in[17] };

    char* ws = (char*)d_ws;
    size_t off = 0;
    auto carve = [&](size_t bytes) -> void* {
        void* p = ws + off;
        off += (bytes + 255) & ~(size_t)255;
        return p;
    };
    _Float16* dtp16   = (_Float16*)carve((size_t)BB * EE * 2);      // 8 MB
    _Float16* wencT   = (_Float16*)carve((size_t)EE * HB * 2);      // 2 MB
    _Float16* wihT    = (_Float16*)carve((size_t)3 * HB * G3 * 2);  // 4.5 MB
    _Float16* whhT    = (_Float16*)carve((size_t)3 * HB * G3 * 2);  // 4.5 MB
    _Float16* x16     = (_Float16*)carve((size_t)BB * HB * 2);      // 2 MB
    float*    giT     = (float*)   carve((size_t)3 * G3 * BB * 4);  // 36 MB

    // 1) precision conversion + weight transposes (k-major for WMMA B operand)
    k_cast16<<<2048, 256, 0, stream>>>(dtp16, dtp, BB * EE);
    k_transpose16<<<1024, 256, 0, stream>>>(wencT, W_enc, HB, EE);
    for (int l = 0; l < 3; ++l) {
        k_transpose16<<<512, 256, 0, stream>>>(wihT + (size_t)l * HB * G3, W_ih[l], G3, HB);
        k_transpose16<<<512, 256, 0, stream>>>(whhT + (size_t)l * HB * G3, W_hh[l], G3, HB);
    }
    // 2) encoder GEMM -> x (f16)
    k_encoder<<<BB / 16, 256, 0, stream>>>(dtp16, wencT, b_enc, x16);
    // 3) step-invariant input gates giT (f32, gate-major transposed)
    dim3 gg(BB / 16, 3);
    k_gi<<<gg, 256, 0, stream>>>(x16, wihT, b_ih[0], b_ih[1], b_ih[2], giT);
    // 4) persistent fused 3-layer GRU over 60 steps + output projection
    k_rnn<<<BB / 16, 256, 0, stream>>>(whhT, giT, b_hh[0], b_hh[1], b_hh[2],
                                       W_out, b_out, (float*)d_out);
}
